// Net_38800734552171
// MI455X (gfx1250) — matrix operand order repaired and checked by reading, not compiled, not run on hardware
//
#include <hip/hip_runtime.h>
#include <hip/hip_bf16.h>
#include <cstdint>

#define EMBED  128
#define HIDDEN 2048
#define G4     (4 * HIDDEN)
#define SEQ    8192
#define VOCAB  128
#define NWG    64
#define SCAN_T 256
#define HEAD_T 512

typedef __attribute__((ext_vector_type(16))) __bf16 v16bf;
typedef __attribute__((ext_vector_type(8)))  float  v8f;

// ---------- bf16 helpers (bit-exact unpack, RNE pack via __bf16 cast) ----------
__device__ __forceinline__ float bf_lo(uint32_t u) {
    uint32_t x = u << 16;
    return __builtin_bit_cast(float, x);
}
__device__ __forceinline__ float bf_hi(uint32_t u) {
    uint32_t x = u & 0xffff0000u;
    return __builtin_bit_cast(float, x);
}

// =====================================================================
// Kernel 1: token tables via WMMA bf16.
//   out[v][j] = sum_k E[v][k] * W[j][k] + bias[j],  v in [0,128), j in [0,J)
// One wave computes one 16x16 tile of D; K=128 -> 4 x v_wmma_f32_16x16x32_bf16.
// A (16x32 bf16): lanes 0-15 row M, elems {K0..7, K16..23}; lanes 16-31 same row,
//                 elems {K8..15, K24..31}.
// B (32x16 bf16): lane n (0-15) = column n, K0..7, K16..23; lane n+16 = column n, K8..15, K24..31.
// D (16x16 f32):  acc[r] -> row r (+8 for lanes>=16), col = lane&15.
// =====================================================================
__global__ void table_wmma_kernel(const float* __restrict__ E,
                                  const float* __restrict__ W,
                                  const float* __restrict__ bias,
                                  float* __restrict__ out, int J)
{
    const int wave    = (blockIdx.x * blockDim.x + threadIdx.x) >> 5;
    const int lane    = threadIdx.x & 31;
    const int tiles_j = J >> 4;
    if (wave >= 8 * tiles_j) return;              // wave-uniform: EXEC stays all-ones

    const int tv = wave / tiles_j, tj = wave % tiles_j;
    const int v0 = tv * 16, j0 = tj * 16;
    const int m  = lane & 15;
    const int hi = lane >> 4;                     // 0 or 1

    v8f acc = {};
    for (int kk = 0; kk < EMBED; kk += 32) {
        v16bf a, b;
        const float* arow = E + (size_t)(v0 + m) * EMBED + kk + hi * 8;
#pragma unroll
        for (int e = 0; e < 8; ++e) {
            a[e]     = (__bf16)arow[e];           // K = base .. base+7
            a[e + 8] = (__bf16)arow[e + 16];      // K = base+16 .. base+23
        }
        const float* brow = W + (size_t)(j0 + m) * EMBED + kk + hi * 8;
#pragma unroll
        for (int e = 0; e < 8; ++e) {
            b[e]     = (__bf16)brow[e];
            b[e + 8] = (__bf16)brow[e + 16];
        }

        acc = __builtin_amdgcn_wmma_f32_16x16x32_bf16(
            /*neg_a=*/false, a, /*neg_b=*/false, b,
            /*c_mod=*/(short)0, acc, /*reuse_a=*/false, /*reuse_b=*/false);
    }
    const float bv = bias[j0 + m];
#pragma unroll
    for (int r = 0; r < 8; ++r) {
        const int row = v0 + r + hi * 8;
        out[(size_t)row * J + (j0 + m)] = acc[r] + bv;
    }
}

// =====================================================================
// Kernel 2: pack f32 weights to bf16 (RNE) so both matrices (~40 MB) stream
// from L2 at half the bytes per scan step.
// =====================================================================
__global__ void f32_to_bf16_kernel(const float* __restrict__ src,
                                   uint16_t* __restrict__ dst, int n)
{
    int i = blockIdx.x * blockDim.x + threadIdx.x;
    const int stride = gridDim.x * blockDim.x;
    for (; i < n; i += stride) {
        __bf16 b = (__bf16)src[i];
        dst[i] = __builtin_bit_cast(uint16_t, b);
    }
}

// =====================================================================
// Kernel 3: zero recurrent state + barrier (fresh every launch -> deterministic)
// =====================================================================
__global__ void init_state_kernel(float* h, float* c, float* m, unsigned* bar)
{
    const int i = blockIdx.x * blockDim.x + threadIdx.x;
    if (i < HIDDEN) { h[i] = 0.f; c[i] = 0.f; m[i] = 0.f; }
    if (i < 2) bar[i] = 0u;
}

// ---------- matvec chunk: W row dot LDS vector ----------
__device__ __forceinline__ float dot_chunk(const uint16_t* __restrict__ wrow,
                                           const float* __restrict__ v,
                                           int kbeg, int klen)
{
    const uint32_t* p = (const uint32_t*)(wrow + kbeg);   // 4B aligned (kbeg even)
    float acc = 0.f;
#pragma unroll 8
    for (int i = 0; i < klen / 2; ++i) {
        const uint32_t u = p[i];
        acc = fmaf(bf_lo(u), v[kbeg + 2 * i],     acc);
        acc = fmaf(bf_hi(u), v[kbeg + 2 * i + 1], acc);
    }
    return acc;
}
__device__ __forceinline__ float dot_chunk(const float* __restrict__ wrow,
                                           const float* __restrict__ v,
                                           int kbeg, int klen)
{
    float acc = 0.f;
#pragma unroll 8
    for (int i = 0; i < klen; ++i) acc = fmaf(wrow[kbeg + i], v[kbeg + i], acc);
    return acc;
}

// ---------- device-wide sense-reversal barrier (agent scope) ----------
__device__ __forceinline__ void grid_barrier(unsigned* cnt, unsigned* gen)
{
    __threadfence();                 // make this thread's stores agent-visible
    __syncthreads();
    if (threadIdx.x == 0) {
        const unsigned g = __hip_atomic_load(gen, __ATOMIC_ACQUIRE,
                                             __HIP_MEMORY_SCOPE_AGENT);
        const unsigned a = __hip_atomic_fetch_add(cnt, 1u, __ATOMIC_ACQ_REL,
                                                  __HIP_MEMORY_SCOPE_AGENT);
        if (a == NWG - 1) {
            __hip_atomic_store(cnt, 0u, __ATOMIC_RELAXED, __HIP_MEMORY_SCOPE_AGENT);
            __hip_atomic_store(gen, g + 1u, __ATOMIC_RELEASE, __HIP_MEMORY_SCOPE_AGENT);
        } else {
            while (__hip_atomic_load(gen, __ATOMIC_ACQUIRE,
                                     __HIP_MEMORY_SCOPE_AGENT) == g)
                __builtin_amdgcn_s_sleep(2);
        }
    }
    __syncthreads();
}

// =====================================================================
// Kernel 4: persistent cooperative scan. 64 WGs x 256 threads.
// WG w owns hidden rows [32w, 32w+32) (and the matching i/f/g/o gate rows);
// c lives in LDS for the whole scan.  Per step:
//   A: m = tm[tok] * (wmh @ h + b)    (32 rows/WG, 8 threads/row)
//   -- grid barrier --
//   B: gates = tg[tok] + wh @ m + b ; c,h update  (128 rows/WG, 2 threads/row)
//   -- grid barrier --
// =====================================================================
template <typename WT>
__global__ void __launch_bounds__(SCAN_T)
scan_kernel(const int* __restrict__ tokens,
            const WT* __restrict__ wh,  const float* __restrict__ wh_b,
            const WT* __restrict__ wmh, const float* __restrict__ wmh_b,
            const float* __restrict__ tg, const float* __restrict__ tm,
            float* __restrict__ hbuf, float* __restrict__ cbuf,
            float* __restrict__ mbuf, unsigned* __restrict__ bar)
{
    __shared__ float vec[HIDDEN];      // staged h (phase A) / m (phase B)
    __shared__ float redA[32 * 8];
    __shared__ float redB[128 * 2];
    __shared__ float gl[128];
    __shared__ float c_loc[32];

    const int tid = threadIdx.x;
    const int wg  = blockIdx.x;
    if (tid < 32) c_loc[tid] = 0.f;

    for (int s = 0; s < SEQ; ++s) {
        const int tok = tokens[s];

        // ---------- Phase A ----------
        for (int i = tid; i < HIDDEN; i += SCAN_T) vec[i] = hbuf[i];
        __syncthreads();
        {
            const int rl = tid >> 3;           // 0..31
            const int ks = tid & 7;            // 0..7  -> K slice of 256
            const int row = wg * 32 + rl;
            redA[tid] = dot_chunk(wmh + (size_t)row * HIDDEN, vec, ks * 256, 256);
        }
        __syncthreads();
        if (tid < 32) {
            const int row = wg * 32 + tid;
            float sum = 0.f;
#pragma unroll
            for (int k = 0; k < 8; ++k) sum += redA[tid * 8 + k];
            mbuf[row] = tm[(size_t)tok * HIDDEN + row] * (sum + wmh_b[row]);
        }
        grid_barrier(bar, bar + 1);

        // ---------- Phase B ----------
        for (int i = tid; i < HIDDEN; i += SCAN_T) vec[i] = mbuf[i];
        __syncthreads();
        {
            const int rl   = tid >> 1;         // 0..127
            const int half = tid & 1;          // K halves of 1024
            const int gate = rl >> 5, jl = rl & 31;
            const int grow = gate * HIDDEN + wg * 32 + jl;
            redB[tid] = dot_chunk(wh + (size_t)grow * HIDDEN, vec, half * 1024, 1024);
        }
        __syncthreads();
        if (tid < 128) {
            const int gate = tid >> 5, jl = tid & 31;
            const int grow = gate * HIDDEN + wg * 32 + jl;
            gl[tid] = redB[tid * 2] + redB[tid * 2 + 1] + wh_b[grow]
                    + tg[(size_t)tok * G4 + grow];
        }
        __syncthreads();
        if (tid < 32) {
            const float iv = gl[tid], fv = gl[32 + tid];
            const float gv = gl[64 + tid], ov = gl[96 + tid];
            const float si = 1.f / (1.f + __expf(-iv));
            const float sf = 1.f / (1.f + __expf(-fv));
            const float so = 1.f / (1.f + __expf(-ov));
            const float cn = sf * c_loc[tid] + si * tanhf(gv);
            c_loc[tid] = cn;
            hbuf[wg * 32 + tid] = so * tanhf(cn);
        }
        grid_barrier(bar, bar + 1);
    }
    if (tid < 32) cbuf[wg * 32 + tid] = c_loc[tid];
}

// =====================================================================
// Kernel 5: MLP head, one workgroup, activations staged in LDS.
// =====================================================================
__global__ void __launch_bounds__(HEAD_T)
head_kernel(const float* __restrict__ hbuf, const float* __restrict__ cbuf,
            const float* __restrict__ fc1w, const float* __restrict__ fc1b,
            const float* __restrict__ fc2w, const float* __restrict__ fc2b,
            const float* __restrict__ fc3w, const float* __restrict__ fc3b,
            const float* __restrict__ fc4w, const float* __restrict__ fc4b,
            float* __restrict__ out)
{
    __shared__ float x0[2 * HIDDEN];
    __shared__ float x1[HIDDEN];
    __shared__ float x2[HIDDEN / 4];
    __shared__ float x3[HIDDEN / 2];
    const int tid = threadIdx.x;

    for (int i = tid; i < HIDDEN; i += HEAD_T) {
        x0[i] = hbuf[i];
        x0[HIDDEN + i] = cbuf[i];
    }
    __syncthreads();
    for (int r = tid; r < HIDDEN; r += HEAD_T) {
        const float* w = fc1w + (size_t)r * (2 * HIDDEN);
        float a = fc1b[r];
        for (int k = 0; k < 2 * HIDDEN; ++k) a = fmaf(w[k], x0[k], a);
        x1[r] = fmaxf(a, 0.f);
    }
    __syncthreads();
    for (int r = tid; r < HIDDEN / 4; r += HEAD_T) {
        const float* w = fc2w + (size_t)r * HIDDEN;
        float a = fc2b[r];
        for (int k = 0; k < HIDDEN; ++k) a = fmaf(w[k], x1[k], a);
        x2[r] = fmaxf(a, 0.f);
    }
    __syncthreads();
    for (int r = tid; r < HIDDEN / 2; r += HEAD_T) {
        const float* w = fc3w + (size_t)r * (HIDDEN / 4);
        float a = fc3b[r];
        for (int k = 0; k < HIDDEN / 4; ++k) a = fmaf(w[k], x2[k], a);
        x3[r] = fmaxf(a, 0.f);
    }
    __syncthreads();
    if (tid < 5) {
        const float* w = fc4w + (size_t)tid * (HIDDEN / 2);
        float a = fc4b[tid];
        for (int k = 0; k < HIDDEN / 2; ++k) a = fmaf(w[k], x3[k], a);
        out[tid] = fmaxf(a, 0.f);
    }
}

// =====================================================================
extern "C" void kernel_launch(void* const* d_in, const int* in_sizes, int n_in,
                              void* d_out, int out_size, void* d_ws, size_t ws_size,
                              hipStream_t stream)
{
    (void)in_sizes; (void)n_in; (void)out_size;
    const int*   tokens  = (const int*)d_in[0];
    const float* embed_w = (const float*)d_in[1];
    const float* wx_w    = (const float*)d_in[2];
    const float* wx_b    = (const float*)d_in[3];
    const float* wh_w    = (const float*)d_in[4];
    const float* wh_b    = (const float*)d_in[5];
    const float* wmx_w   = (const float*)d_in[6];
    const float* wmx_b   = (const float*)d_in[7];
    const float* wmh_w   = (const float*)d_in[8];
    const float* wmh_b   = (const float*)d_in[9];
    const float* fc1w = (const float*)d_in[10];
    const float* fc1b = (const float*)d_in[11];
    const float* fc2w = (const float*)d_in[12];
    const float* fc2b = (const float*)d_in[13];
    const float* fc3w = (const float*)d_in[14];
    const float* fc3b = (const float*)d_in[15];
    const float* fc4w = (const float*)d_in[16];
    const float* fc4b = (const float*)d_in[17];

    char* ws = (char*)d_ws;
    size_t off = 0;
    auto carve = [&](size_t bytes) -> void* {
        void* p = ws + off;
        off = (off + bytes + 255) & ~(size_t)255;
        return p;
    };
    float*    tg    = (float*)carve((size_t)VOCAB * G4 * 4);       // 4 MB
    float*    tm    = (float*)carve((size_t)VOCAB * HIDDEN * 4);   // 1 MB
    float*    hb    = (float*)carve(HIDDEN * 4);
    float*    cb    = (float*)carve(HIDDEN * 4);
    float*    mb    = (float*)carve(HIDDEN * 4);
    unsigned* bar   = (unsigned*)carve(256);
    uint16_t* wh16  = (uint16_t*)carve((size_t)G4 * HIDDEN * 2);   // 32 MB
    uint16_t* wmh16 = (uint16_t*)carve((size_t)HIDDEN * HIDDEN * 2); // 8 MB
    const bool use_bf16 = (ws_size >= off);

    // token tables via WMMA: 8 v-tiles x (J/16) j-tiles, 8 waves per block
    table_wmma_kernel<<<(8 * (G4 / 16)) / 8,     256, 0, stream>>>(embed_w, wx_w,  wx_b,  tg, G4);
    table_wmma_kernel<<<(8 * (HIDDEN / 16)) / 8, 256, 0, stream>>>(embed_w, wmx_w, wmx_b, tm, HIDDEN);
    init_state_kernel<<<(HIDDEN + 255) / 256, 256, 0, stream>>>(hb, cb, mb, bar);

    if (use_bf16) {
        f32_to_bf16_kernel<<<2048, 256, 0, stream>>>(wh_w,  wh16,  G4 * HIDDEN);
        f32_to_bf16_kernel<<<512,  256, 0, stream>>>(wmh_w, wmh16, HIDDEN * HIDDEN);
        scan_kernel<uint16_t><<<NWG, SCAN_T, 0, stream>>>(
            tokens, wh16, wh_b, wmh16, wmh_b, tg, tm, hb, cb, mb, bar);
    } else {
        scan_kernel<float><<<NWG, SCAN_T, 0, stream>>>(
            tokens, wh_w, wh_b, wmh_w, wmh_b, tg, tm, hb, cb, mb, bar);
    }

    head_kernel<<<1, HEAD_T, 0, stream>>>(hb, cb, fc1w, fc1b, fc2w, fc2b,
                                          fc3w, fc3b, fc4w, fc4b, (float*)d_out);
}
